// Memory_69715909149128
// MI455X (gfx1250) — compile-verified, matrix operand order checked by reading, not run on hardware
//
#include <hip/hip_runtime.h>
#include <hip/hip_bf16.h>
#include <math.h>

// ---------------- problem constants ----------------
#define BQ   1024      // batch
#define DD   128       // feature dim
#define MM   262144    // memory slots
#define KK   256       // top-k
#define NT   128       // keys per chunk in the score kernel
#define TEMPC   1.0f   // max(1, log(0.2*256)/40) == 1.0
#define MARGINC 0.1f

// d_out layout (float elements, concatenated reference outputs)
#define OUT_YHAT 0ull
#define OUT_SOFT 1024ull
#define OUT_LOSS (OUT_SOFT + 1024ull*256ull)          // 263168
#define OUT_KEYS (OUT_LOSS + 1ull)                    // 263169
#define OUT_VALS (OUT_KEYS + 262144ull*128ull)        // 33817601
#define OUT_AGE  (OUT_VALS + 262144ull)               // 34079745

typedef __attribute__((ext_vector_type(16))) __bf16 v16bf;
typedef __attribute__((ext_vector_type(8)))  float  v8f;
typedef __attribute__((ext_vector_type(4)))  unsigned int v4u;
typedef __attribute__((ext_vector_type(8)))  unsigned int v8u;

__device__ __forceinline__ unsigned short f32_to_bf16_rne(float f) {
  unsigned int u = __float_as_uint(f);
  u += 0x7fffu + ((u >> 16) & 1u);
  return (unsigned short)(u >> 16);
}
__device__ __forceinline__ __bf16 us_to_bf16(unsigned short s) {
  union { unsigned short u; __bf16 b; } cv; cv.u = s; return cv.b;
}

// ---- CDNA5 async memory->LDS DMA (ASYNCcnt-tracked) ----
// GVS mode: mem_addr = SGPR_base + VGPR_i32_offset ; per-lane 16B to LDS.
__device__ __forceinline__ void async_load_b128(unsigned lds_addr, int goff,
                                                const void* sbase) {
  asm volatile("global_load_async_to_lds_b128 %0, %1, %2"
               :: "v"(lds_addr), "v"(goff), "s"(sbase)
               : "memory");
}
__device__ __forceinline__ void wait_async_le32() {
  asm volatile("s_wait_asynccnt 0x20" ::: "memory");
}
__device__ __forceinline__ void wait_async_0() {
  asm volatile("s_wait_asynccnt 0x0" ::: "memory");
}

// ---- CDNA5 Tensor Data Mover (TENSORcnt-tracked) ----
// D# group 0: count=1 | lds_addr | global_addr(57b) | type=2 ("image")
__device__ __forceinline__ v4u tdm_group0(unsigned lds_addr, const void* gptr) {
  const unsigned long long ga = (unsigned long long)gptr;
  v4u g;
  g[0] = 1u;                                    // count=1 valid, user mode
  g[1] = lds_addr;                              // bits[63:32]  LDS byte addr
  g[2] = (unsigned)ga;                          // global_addr[31:0]
  g[3] = (unsigned)((ga >> 32) & 0x01ffffffu)   // global_addr[56:32]
         | (2u << 30);                          // type = 2
  return g;
}
// D# group 1 for a dense 2D tile (tile == tensor): data_size=4B,
// tensor_dim0=tile_dim0=d0, tensor_dim1=tile_dim1=d1, dim0 stride=stride0.
__device__ __forceinline__ v8u tdm_group1_2d(unsigned d0, unsigned d1,
                                             unsigned stride0) {
  v8u g;
  g[0] = 2u << 16;                              // data_size = 4 bytes, mask=0
  g[1] = (d0 & 0xffffu) << 16;                  // tensor_dim0[15:0]
  g[2] = (d0 >> 16) | ((d1 & 0xffffu) << 16);   // tensor_dim0[31:16]|dim1[15:0]
  g[3] = (d1 >> 16) | ((d0 & 0xffffu) << 16);   // dim1[31:16] | tile_dim0
  g[4] = d1 & 0xffffu;                          // tile_dim1 | tile_dim2=0
  g[5] = stride0;                               // tensor_dim0_stride[31:0]
  g[6] = 0u;                                    // stride0[47:32] | dim1_stride lo
  g[7] = 0u;
  return g;
}
__device__ __forceinline__ void tdm_load(v4u g0, v8u g1) {
  asm volatile("tensor_load_to_lds %0, %1" :: "s"(g0), "s"(g1) : "memory");
}
__device__ __forceinline__ void tdm_store(v4u g0, v8u g1) {
  asm volatile("tensor_store_from_lds %0, %1" :: "s"(g0), "s"(g1) : "memory");
}

// ---------------- K0: zero small accumulators ----------------
__global__ void init_misc_kernel(int* inc_cnt, float* loss_acc) {
  if (threadIdx.x == 0) { *inc_cnt = 0; *loss_acc = 0.0f; }
}

// ---------------- K1: L2-normalize queries ----------------
__global__ __launch_bounds__(128) void normalize_kernel(
    const float* __restrict__ x, float* __restrict__ qf,
    unsigned short* __restrict__ qb) {
  __shared__ float red[128];
  const int b = blockIdx.x, t = threadIdx.x;
  float v = x[(size_t)b * DD + t];
  red[t] = v * v;
  __syncthreads();
  for (int s = 64; s > 0; s >>= 1) {
    if (t < s) red[t] += red[t + s];
    __syncthreads();
  }
  const float inv = 1.0f / fmaxf(sqrtf(red[0]), 1e-12f);
  const float q = v * inv;
  qf[(size_t)b * DD + t] = q;
  qb[(size_t)b * DD + t] = f32_to_bf16_rne(q);
}

// ---------------- K2: fused WMMA scores + streaming top-K ----------------
// 64 blocks x 128 threads; each block owns 16 query rows, streams all M keys
// through an async-DMA double-buffered LDS pipeline.
#define SMEM_STAGE 0u
#define SMEM_K     131072u
#define SMEM_Q     163840u
#define SMEM_S     167936u
#define SMEM_TV    176128u
#define SMEM_TI    192512u
#define SMEM_RS    208896u
#define SMEM_BYTES 208960u
#define CHUNK_BYTES (NT * DD * 4)     // 65536

__global__ __launch_bounds__(128) void score_topk_kernel(
    const unsigned short* __restrict__ qb, const float* __restrict__ keys,
    const int* __restrict__ values, const int* __restrict__ y,
    float* __restrict__ out, float* loss_acc, int* inc_cnt,
    int* __restrict__ ws_yidx, int* __restrict__ ws_res) {
  extern __shared__ __align__(16) char smem[];
  float*  stage = (float*)(smem + SMEM_STAGE);
  __bf16* sh_k  = (__bf16*)(smem + SMEM_K);
  __bf16* sh_q  = (__bf16*)(smem + SMEM_Q);
  float*  sh_s  = (float*)(smem + SMEM_S);
  float*  sh_tv = (float*)(smem + SMEM_TV);
  int*    sh_ti = (int*)(smem + SMEM_TI);
  float*  sh_rsum = (float*)(smem + SMEM_RS);

  const int tid  = threadIdx.x;
  const int lane = tid & 31;
  const int wave = tid >> 5;          // 4 waves
  const int half = lane >> 4;         // 0 / 1
  const int l15  = lane & 15;
  const int rowbase = blockIdx.x * 16;

  // low 32 bits of a generic LDS pointer == LDS byte address for DS/async ops
  const unsigned stage_lo = (unsigned)(unsigned long long)(void*)stage;

  // load 16xD query tile as bf16
  for (int i = tid; i < 16 * DD; i += 128)
    sh_q[i] = us_to_bf16(qb[(size_t)rowbase * DD + i]);
  // init top-K buffers
  for (int i = tid; i < 16 * KK; i += 128) {
    sh_tv[i] = -1e30f;
    sh_ti[i] = 0;
  }

  // prologue: async-DMA chunk 0 into stage[0] (32 x b128 per thread)
  {
    const unsigned l0 = stage_lo + tid * 16;
    const int g0 = tid * 16;
    for (int j = 0; j < 32; ++j)
      async_load_b128(l0 + j * 2048, g0 + j * 2048, keys);
  }
  __syncthreads();

  // Hoist A fragments (16x32 bf16 per k-step, ISA lane layout):
  // lane<16: row=lane,  K in {k0..k0+7} U {k0+16..k0+23}; lane>=16: K +8
  v16bf afr[4];
  for (int ks = 0; ks < 4; ++ks) {
    const __bf16* qr = &sh_q[l15 * DD + ks * 32 + half * 8];
    for (int j = 0; j < 8; ++j) { afr[ks][j] = qr[j]; afr[ks][j + 8] = qr[16 + j]; }
  }

  float rmin = -1e30f;  // running K-th value (valid for tid < 16)
  int   rminpos = 0;

  const int nchunks = MM / NT;        // 2048
  for (int c = 0; c < nchunks; ++c) {
    const int buf = c & 1;
    // issue DMA for next chunk into the other stage buffer, then wait for
    // the current chunk only (<=32 leaves the new transfers in flight)
    if (c + 1 < nchunks) {
      const unsigned l0 = stage_lo + (unsigned)(buf ^ 1) * CHUNK_BYTES + tid * 16;
      const int g0 = (c + 1) * CHUNK_BYTES + tid * 16;
      for (int j = 0; j < 32; ++j)
        async_load_b128(l0 + j * 2048, g0 + j * 2048, keys);
      wait_async_le32();
    } else {
      wait_async_0();
    }
    // L2 warm for chunk c+2
    if (c + 2 < nchunks) {
      const char* nxt = (const char*)keys + (size_t)(c + 2) * CHUNK_BYTES;
      for (int p = tid; p < 256; p += 128)
        __builtin_prefetch(nxt + p * 256, 0, 1);
    }
    __syncthreads();

    // convert staged f32 -> bf16 tile (LDS -> LDS)
    {
      const float4* sb = (const float4*)(stage + (size_t)buf * NT * DD);
      uint2* kb = (uint2*)sh_k;
      for (int i = tid; i < NT * DD / 4; i += 128) {
        const float4 v4 = sb[i];
        uint2 p;
        p.x = (unsigned)f32_to_bf16_rne(v4.x) | ((unsigned)f32_to_bf16_rne(v4.y) << 16);
        p.y = (unsigned)f32_to_bf16_rne(v4.z) | ((unsigned)f32_to_bf16_rne(v4.w) << 16);
        kb[i] = p;
      }
    }
    __syncthreads();

    // Each wave computes two 16x16 tiles: ntile = wave*2 + {0,1}
    for (int tt = 0; tt < 2; ++tt) {
      const int ntile = wave * 2 + tt;
      v8f acc = {};
      const int nrow = ntile * 16 + l15;   // B operand: key row == out column
      for (int ks = 0; ks < 4; ++ks) {
        v16bf bfr;
        const __bf16* kr = &sh_k[nrow * DD + ks * 32 + half * 8];
        for (int j = 0; j < 8; ++j) { bfr[j] = kr[j]; bfr[j + 8] = kr[16 + j]; }
        acc = __builtin_amdgcn_wmma_f32_16x16x32_bf16(
            false, afr[ks], false, bfr, (short)0, acc, false, false);
      }
      // C layout: VGPR r -> (M = r + half*8, N = l15)
      for (int r = 0; r < 8; ++r)
        sh_s[(r + half * 8) * NT + ntile * 16 + l15] = acc[r];
    }
    __syncthreads();

    // merge chunk scores into running top-K (one thread per row)
    if (tid < 16) {
      const int gb = c * NT;
      float* tv = &sh_tv[tid * KK];
      int*   ti = &sh_ti[tid * KK];
      const float* srow = &sh_s[tid * NT];
      for (int j = 0; j < NT; ++j) {
        const float v = srow[j];
        if (v > rmin) {
          tv[rminpos] = v; ti[rminpos] = gb + j;
          float m = tv[0]; int mp = 0;
          for (int q = 1; q < KK; ++q) { float xq = tv[q]; if (xq < m) { m = xq; mp = q; } }
          rmin = m; rminpos = mp;
        }
      }
    }
    __syncthreads();
  }

  // bitonic sort top-K per row, descending by value (tie: lower index first)
  for (int size = 2; size <= KK; size <<= 1)
    for (int stride = size >> 1; stride > 0; stride >>= 1) {
      __syncthreads();
      const int row = tid >> 3, tr = tid & 7;
      for (int q = 0; q < 16; ++q) {
        const int i  = tr * 16 + q;                       // CE id 0..127
        const int lo = ((i & ~(stride - 1)) << 1) | (i & (stride - 1));
        const int hi = lo + stride;
        const bool desc = ((lo & size) == 0);
        float v1 = sh_tv[row * KK + lo], v2 = sh_tv[row * KK + hi];
        int   i1 = sh_ti[row * KK + lo], i2 = sh_ti[row * KK + hi];
        const bool sw = desc ? (v1 < v2 || (v1 == v2 && i1 > i2))
                             : (v1 > v2 || (v1 == v2 && i1 < i2));
        if (sw) {
          sh_tv[row * KK + lo] = v2; sh_tv[row * KK + hi] = v1;
          sh_ti[row * KK + lo] = i2; sh_ti[row * KK + hi] = i1;
        }
      }
    }
  __syncthreads();

  // softmax denominators
  if (tid < 16) {
    const float mx = sh_tv[tid * KK + 0] * TEMPC;
    float s = 0.0f;
    for (int k = 0; k < KK; ++k) s += __expf(sh_tv[tid * KK + k] * TEMPC - mx);
    sh_rsum[tid] = s;
  }
  __syncthreads();
  // write softmax scores
  for (int i = tid; i < 16 * KK; i += 128) {
    const int row = i >> 8, k = i & 255;
    const float mx = sh_tv[row * KK + 0] * TEMPC;
    out[OUT_SOFT + (size_t)(rowbase + row) * KK + k] =
        __expf(sh_tv[row * KK + k] * TEMPC - mx) / sh_rsum[row];
  }
  // per-row: y_hat, result flag, margin loss
  if (tid < 16) {
    const int grow = rowbase + tid;
    const int yv   = y[grow];
    const int idx0 = sh_ti[tid * KK + 0];
    const int yhat = values[idx0];
    out[OUT_YHAT + grow] = (float)yhat;
    ws_yidx[grow] = idx0;
    const int res = (yhat == yv) ? 1 : 0;
    ws_res[grow] = res;
    if (!res) atomicAdd(inc_cnt, 1);
    float pos = 0.0f, neg = 0.0f;
    for (int k = 0; k < KK; ++k) {
      const float cosv = sh_tv[tid * KK + k];
      const float msk  = (values[sh_ti[tid * KK + k]] == yv) ? 1.0f : 0.0f;
      const float a = cosv * msk, b2 = cosv * (1.0f - msk);
      if (k == 0) { pos = a; neg = b2; }
      else        { pos = fmaxf(pos, a); neg = fmaxf(neg, b2); }
    }
    atomicAdd(loss_acc, fmaxf(neg - pos + MARGINC, 0.0f));
  }
}

// ---------------- K3a: new_keys = keys_m via Tensor Data Mover ----------------
// 256 single-wave workgroups; each streams 1024 rows in 16 double-buffered
// 64x128-f32 tiles: tensor_load_to_lds(c+1) overlaps tensor_store_from_lds(c).
#define TDM_ROWS 64
#define TDM_CHUNK_BYTES (TDM_ROWS * DD * 4)   // 32768
#define TDM_WGS 256
__global__ __launch_bounds__(32) void tdm_copy_keys_kernel(
    const float* __restrict__ keys, float* __restrict__ out) {
  extern __shared__ __align__(16) char smem[];
  const unsigned lds_lo = (unsigned)(unsigned long long)(void*)smem;
  const size_t row0 = (size_t)blockIdx.x * (MM / TDM_WGS);   // 1024 rows/WG
  const float* src = keys + row0 * DD;
  float* dst = out + OUT_KEYS + row0 * DD;
  const int nch = (MM / TDM_WGS) / TDM_ROWS;                 // 16 chunks
  const v8u g1 = tdm_group1_2d(DD, TDM_ROWS, DD);
  // prologue: load chunk 0 into buffer 0
  tdm_load(tdm_group0(lds_lo, src), g1);
  for (int c = 0; c < nch; ++c) {
    const int buf = c & 1;
    __builtin_amdgcn_s_wait_tensorcnt(0);       // chunk c landed; prior store done
    if (c + 1 < nch)
      tdm_load(tdm_group0(lds_lo + (unsigned)(buf ^ 1) * TDM_CHUNK_BYTES,
                          src + (size_t)(c + 1) * TDM_ROWS * DD), g1);
    tdm_store(tdm_group0(lds_lo + (unsigned)buf * TDM_CHUNK_BYTES,
                         dst + (size_t)c * TDM_ROWS * DD), g1);
  }
  __builtin_amdgcn_s_wait_tensorcnt(0);
}

// ---------------- K3b: values (as float), age+1 ----------------
__global__ void vals_age_kernel(const int* __restrict__ values,
                                const float* __restrict__ age,
                                float* __restrict__ out) {
  const int gid = blockIdx.x * blockDim.x + threadIdx.x;
  if (gid < MM) {
    out[OUT_VALS + gid] = (float)values[gid];
    out[OUT_AGE  + gid] = age[gid] + 1.0f;
  }
}

// ---------------- K4: correct-prediction key update ----------------
__global__ __launch_bounds__(128) void correct_update_kernel(
    const float* __restrict__ keys, const float* __restrict__ qf,
    const int* __restrict__ ws_res, const int* __restrict__ ws_yidx,
    float* __restrict__ out) {
  __shared__ float red[128];
  const int b = blockIdx.x;
  if (!ws_res[b]) return;            // uniform per block
  const int slot = ws_yidx[b];
  const int t = threadIdx.x;
  const float v = keys[(size_t)slot * DD + t] + qf[(size_t)b * DD + t];
  red[t] = v * v;
  __syncthreads();
  for (int s = 64; s > 0; s >>= 1) {
    if (t < s) red[t] += red[t + s];
    __syncthreads();
  }
  const float inv = 1.0f / fmaxf(sqrtf(red[0]), 1e-12f);
  out[OUT_KEYS + (size_t)slot * DD + t] = v * inv;
  if (t == 0) out[OUT_AGE + slot] = 0.0f;
}

// ---------------- K5: exact top-B oldest via histogram select ----------------
#define NB     2048
#define TIECAP 2048
#define SORTN  4096
__global__ __launch_bounds__(1024) void select_oldest_kernel(
    const float* __restrict__ out_ro, const float* __restrict__ noise,
    float* __restrict__ l1v, int* __restrict__ l1i,
    float* __restrict__ l2v, int* __restrict__ l2i,
    int* __restrict__ oldest) {
  __shared__ int   hist[NB];
  __shared__ float sv[SORTN];
  __shared__ int   si[SORTN];
  __shared__ int   s_bt, s_c1, s_c2;
  const int t = threadIdx.x;
  for (int i = t; i < NB; i += 1024) hist[i] = 0;
  if (t == 0) { s_c1 = 0; s_c2 = 0; s_bt = 0; }
  __syncthreads();
  const float VMIN = -32.0f, SCALE = (float)NB / 64.0f;
  for (int i = t; i < MM; i += 1024) {
    const float v = out_ro[OUT_AGE + i] + noise[i];
    int bin = (int)((v - VMIN) * SCALE);
    bin = bin < 0 ? 0 : (bin > NB - 1 ? NB - 1 : bin);
    atomicAdd(&hist[bin], 1);
  }
  __syncthreads();
  if (t == 0) {
    int cum = 0, bt = 0;
    for (int bb = NB - 1; bb >= 0; --bb) {
      cum += hist[bb];
      if (cum >= BQ) { bt = bb; break; }
    }
    s_bt = bt;
  }
  __syncthreads();
  const int bt = s_bt;
  for (int i = t; i < MM; i += 1024) {
    const float v = out_ro[OUT_AGE + i] + noise[i];
    int bin = (int)((v - VMIN) * SCALE);
    bin = bin < 0 ? 0 : (bin > NB - 1 ? NB - 1 : bin);
    if (bin > bt) {
      const int p = atomicAdd(&s_c1, 1);
      if (p < BQ) { l1v[p] = v; l1i[p] = i; }
    } else if (bin == bt) {
      const int p = atomicAdd(&s_c2, 1);
      if (p < TIECAP) { l2v[p] = v; l2i[p] = i; }
    }
  }
  __syncthreads();
  const int c1 = s_c1 < BQ ? s_c1 : BQ;
  const int c2 = s_c2 < TIECAP ? s_c2 : TIECAP;
  for (int i = t; i < SORTN; i += 1024) {
    if (i < c1)            { sv[i] = l1v[i];      si[i] = l1i[i]; }
    else if (i < c1 + c2)  { sv[i] = l2v[i - c1]; si[i] = l2i[i - c1]; }
    else                   { sv[i] = -1e30f;      si[i] = 0x7fffffff; }
  }
  for (int size = 2; size <= SORTN; size <<= 1)
    for (int stride = size >> 1; stride > 0; stride >>= 1) {
      __syncthreads();
      for (int i = t; i < SORTN / 2; i += 1024) {
        const int lo = ((i & ~(stride - 1)) << 1) | (i & (stride - 1));
        const int hi = lo + stride;
        const bool desc = ((lo & size) == 0);
        float v1 = sv[lo], v2 = sv[hi];
        int   i1 = si[lo], i2 = si[hi];
        const bool sw = desc ? (v1 < v2 || (v1 == v2 && i1 > i2))
                             : (v1 > v2 || (v1 == v2 && i1 < i2));
        if (sw) { sv[lo] = v2; sv[hi] = v1; si[lo] = i2; si[hi] = i1; }
      }
    }
  __syncthreads();
  if (t < BQ) oldest[t] = si[t];
}

// ---------------- K6: insert new memories at oldest slots ----------------
__global__ __launch_bounds__(128) void insert_kernel(
    const float* __restrict__ qf, const int* __restrict__ y,
    const int* __restrict__ oldest, const int* __restrict__ inc_cnt,
    float* __restrict__ out) {
  if (*inc_cnt == 0) return;
  const int b = blockIdx.x, t = threadIdx.x;
  const int slot = oldest[b];
  out[OUT_KEYS + (size_t)slot * DD + t] = qf[(size_t)b * DD + t];
  if (t == 0) {
    out[OUT_VALS + slot] = (float)y[b];
    out[OUT_AGE  + slot] = 0.0f;
  }
}

// ---------------- K7: finalize loss ----------------
__global__ void finalize_kernel(const float* loss_acc, float* out) {
  if (threadIdx.x == 0 && blockIdx.x == 0)
    out[OUT_LOSS] = *loss_acc / (float)BQ;
}

// ---------------- host launcher ----------------
extern "C" void kernel_launch(void* const* d_in, const int* in_sizes, int n_in,
                              void* d_out, int out_size, void* d_ws, size_t ws_size,
                              hipStream_t stream) {
  const float* x      = (const float*)d_in[0];
  const int*   y      = (const int*)  d_in[1];
  const float* keys   = (const float*)d_in[2];
  const int*   values = (const int*)  d_in[3];
  const float* age    = (const float*)d_in[4];
  const float* noise  = (const float*)d_in[5];
  float* out = (float*)d_out;
  char*  ws  = (char*)d_ws;

  float*          qf       = (float*)(ws + 0);               // B*D f32  (512 KB)
  unsigned short* qb       = (unsigned short*)(ws + 524288); // B*D bf16 (256 KB)
  int*            ws_yidx  = (int*)(ws + 786432);            // B ints
  int*            ws_res   = (int*)(ws + 790528);            // B ints
  int*            inc_cnt  = (int*)(ws + 794624);
  float*          loss_acc = (float*)(ws + 794628);
  float*          l1v      = (float*)(ws + 794688);
  int*            l1i      = (int*)(ws + 798784);
  float*          l2v      = (float*)(ws + 802880);
  int*            l2i      = (int*)(ws + 811072);
  int*            oldest   = (int*)(ws + 819264);

  init_misc_kernel<<<1, 32, 0, stream>>>(inc_cnt, loss_acc);
  normalize_kernel<<<BQ, 128, 0, stream>>>(x, qf, qb);
  score_topk_kernel<<<BQ / 16, 128, SMEM_BYTES, stream>>>(
      qb, keys, values, y, out, loss_acc, inc_cnt, ws_yidx, ws_res);
  tdm_copy_keys_kernel<<<TDM_WGS, 32, 2 * TDM_CHUNK_BYTES, stream>>>(keys, out);
  vals_age_kernel<<<MM / 256, 256, 0, stream>>>(values, age, out);
  correct_update_kernel<<<BQ, 128, 0, stream>>>(keys, qf, ws_res, ws_yidx, out);
  select_oldest_kernel<<<1, 1024, 0, stream>>>(out, noise, l1v, l1i, l2v, l2i, oldest);
  insert_kernel<<<BQ, 128, 0, stream>>>(qf, y, oldest, inc_cnt, out);
  finalize_kernel<<<1, 32, 0, stream>>>(loss_acc, out);
}
